// multi_HAN_48000554500224
// MI455X (gfx1250) — matrix operand order repaired and checked by reading, not compiled, not run on hardware
//
#include <hip/hip_runtime.h>

// ---------------------------------------------------------------------------
// multi_HAN on MI455X (gfx1250): 10 fused [1024,K]@[K,256] GEMMs on the
// bf16 WMMA path (f32 accumulate) with a double-buffered LDS pipeline,
// then fused VALU kernels for homo/hete attention and the final softmax.
// Block tile M=64, N=64 (128 threads / 4 waves) for fine-grained WGP packing.
// ---------------------------------------------------------------------------

typedef __bf16  v16bf __attribute__((ext_vector_type(16)));
typedef __bf16  v2bf  __attribute__((ext_vector_type(2)));
typedef float   v8f   __attribute__((ext_vector_type(8)));

struct __attribute__((aligned(16))) B128v { unsigned int a, b, c, d; };
struct __attribute__((aligned(8)))  B64v  { unsigned int a, b; };
union FragBF { unsigned int u[8]; v16bf bf; };

// Scalar f32->bf16, round-half-up (tail path only)
__device__ __forceinline__ unsigned short f2bf(float f) {
    return (unsigned short)((__float_as_uint(f) + 0x8000u) >> 16);
}

// Pack two f32 -> packed bf16x2. Prefer HW cvt_pk if the builtin exists;
// fallback is round-half-up (differs from RNE only at exact tie midpoints).
__device__ __forceinline__ unsigned int pack2bf(float lo, float hi) {
#if __has_builtin(__builtin_amdgcn_cvt_pk_bf16_f32)
    v2bf r = __builtin_amdgcn_cvt_pk_bf16_f32(lo, hi);
    return __builtin_bit_cast(unsigned int, r);
#else
    unsigned int a = (__float_as_uint(lo) + 0x8000u) >> 16;
    unsigned int b = (__float_as_uint(hi) + 0x8000u) & 0xFFFF0000u;
    return a | b;
#endif
}

// ---------------------------------------------------------------------------
struct GemmJob   { const float* X; const float* W; const float* bias; float* out; int K; };
struct GemmBatch { GemmJob j[10]; };

// Stage A chunk: X[mb..mb+63][k0..k0+31] -> la (bf16 [m][k]), 128 threads
__device__ __forceinline__ void stage_A(const float* __restrict__ X, int K, int mb, int k0,
                                        int t, unsigned short* __restrict__ la, bool full) {
    const int r  = t >> 1;            // 0..63
    const int cb = (t & 1) * 16;      // 0 or 16
    const float* src = X + (size_t)(mb + r) * K + k0 + cb;
    unsigned short* drow = &la[r * 32 + cb];
    if (full) {
        __builtin_prefetch((const void*)(src + 64), 0, 1);   // two chunks ahead
        const float4* s4 = (const float4*)src;
        float4 v0 = s4[0], v1 = s4[1], v2 = s4[2], v3 = s4[3];
        B128v o0 = { pack2bf(v0.x, v0.y), pack2bf(v0.z, v0.w),
                     pack2bf(v1.x, v1.y), pack2bf(v1.z, v1.w) };
        B128v o1 = { pack2bf(v2.x, v2.y), pack2bf(v2.z, v2.w),
                     pack2bf(v3.x, v3.y), pack2bf(v3.z, v3.w) };
        *(B128v*)(drow)     = o0;
        *(B128v*)(drow + 8) = o1;
    } else {
        #pragma unroll
        for (int e = 0; e < 16; ++e) {
            const int kk = k0 + cb + e;
            drow[e] = f2bf(kk < K ? src[e] : 0.0f);
        }
    }
}

// Stage B chunk transposed: W[k0..k0+31][nb..nb+63] -> lb (bf16 [n][k]), 128 threads
__device__ __forceinline__ void stage_B(const float* __restrict__ W, int K, int nb, int k0,
                                        int t, unsigned short* __restrict__ lb, bool full) {
    const int kr = (t >> 4) * 4;      // 0,4,...,28 -> k rows kr..kr+3
    const int n4 = (t & 15) * 4;      // 0,4,...,60
    const float* w0 = W + (size_t)(k0 + kr) * 256 + nb + n4;
    float4 a, b, c, d;
    if (full) {
        __builtin_prefetch((const void*)(w0 + (size_t)64 * 256), 0, 1);
        a = *(const float4*)(w0);
        b = *(const float4*)(w0 + 256);
        c = *(const float4*)(w0 + 512);
        d = *(const float4*)(w0 + 768);
    } else {
        a.x=a.y=a.z=a.w=0.f; b.x=b.y=b.z=b.w=0.f;
        c.x=c.y=c.z=c.w=0.f; d.x=d.y=d.z=d.w=0.f;
        if (k0 + kr + 0 < K) { a = *(const float4*)(w0); }
        if (k0 + kr + 1 < K) { b = *(const float4*)(w0 + 256); }
        if (k0 + kr + 2 < K) { c = *(const float4*)(w0 + 512); }
        if (k0 + kr + 3 < K) { d = *(const float4*)(w0 + 768); }
    }
    B64v s0 = { pack2bf(a.x, b.x), pack2bf(c.x, d.x) };
    B64v s1 = { pack2bf(a.y, b.y), pack2bf(c.y, d.y) };
    B64v s2 = { pack2bf(a.z, b.z), pack2bf(c.z, d.z) };
    B64v s3 = { pack2bf(a.w, b.w), pack2bf(c.w, d.w) };
    *(B64v*)&lb[(n4 + 0) * 32 + kr] = s0;
    *(B64v*)&lb[(n4 + 1) * 32 + kr] = s1;
    *(B64v*)&lb[(n4 + 2) * 32 + kr] = s2;
    *(B64v*)&lb[(n4 + 3) * 32 + kr] = s3;
}

__device__ __forceinline__ void compute_chunk(const unsigned short* __restrict__ la,
                                              const unsigned short* __restrict__ lb,
                                              int wave, int l16, int hlf, v8f acc[4]) {
    // A fragment loaded once; B fragments per tile; WMMAs issued back-to-back.
    FragBF fa, fb[4];
    const B128v* rowA = (const B128v*)&la[(wave * 16 + l16) * 32];
    *(B128v*)&fa.u[0] = rowA[hlf];        // lanes 0-15: K0-7 ; lanes 16-31: K8-15
    *(B128v*)&fa.u[4] = rowA[2 + hlf];    // lanes 0-15: K16-23; lanes 16-31: K24-31
    #pragma unroll
    for (int tt = 0; tt < 4; ++tt) {
        const B128v* rowB = (const B128v*)&lb[(tt * 16 + l16) * 32];
        *(B128v*)&fb[tt].u[0] = rowB[hlf];          // lanes 0-15: K0-7 ; 16-31: K8-15
        *(B128v*)&fb[tt].u[4] = rowB[2 + hlf];      // lanes 0-15: K16-23; 16-31: K24-31
    }
    #pragma unroll
    for (int tt = 0; tt < 4; ++tt) {
        acc[tt] = __builtin_amdgcn_wmma_f32_16x16x32_bf16(
            false, fa.bf, false, fb[tt].bf, (short)0, acc[tt], false, false);
    }
}

__global__ __launch_bounds__(128)
void embed_gemm(GemmBatch jb) {
    const GemmJob job = jb.j[blockIdx.z];
    const int K  = job.K;
    const int mb = blockIdx.y * 64;
    const int nb = blockIdx.x * 64;

    const int t    = threadIdx.x;
    const int lane = t & 31;
    const int wave = t >> 5;        // 0..3
    const int hlf  = lane >> 4;
    const int l16  = lane & 15;

    __shared__ unsigned short lds_a [2][64 * 32];    // 2 x 4 KB
    __shared__ unsigned short lds_bt[2][64 * 32];    // 2 x 4 KB

    v8f acc[4] = {};

    const int nch = (K + 31) >> 5;
    stage_A(job.X, K, mb, 0, t, lds_a[0],  32 <= K);
    stage_B(job.W, K, nb, 0, t, lds_bt[0], 32 <= K);

    for (int c = 0; c < nch; ++c) {
        __syncthreads();
        if (c + 1 < nch) {
            const int  k1   = (c + 1) << 5;
            const bool full = (k1 + 32 <= K);
            stage_A(job.X, K, mb, k1, t, lds_a [(c + 1) & 1], full);
            stage_B(job.W, K, nb, k1, t, lds_bt[(c + 1) & 1], full);
        }
        compute_chunk(lds_a[c & 1], lds_bt[c & 1], wave, l16, hlf, acc);
    }

    // epilogue: bias add + store (C/D layout: VGPR r -> M = r + 8*hlf, N = lane&15)
    #pragma unroll
    for (int tt = 0; tt < 4; ++tt) {
        const int   n  = nb + tt * 16 + l16;
        const float bv = job.bias[n];
        #pragma unroll
        for (int r = 0; r < 8; ++r) {
            const int m = mb + wave * 16 + r + hlf * 8;
            job.out[(size_t)m * 256 + n] = acc[tt][r] + bv;
        }
    }
}

// ---------------------------------------------------------------------------
// Per-facet fixed-order tree reduction (deterministic). One block = one batch
// row, 256 threads, F=4 facets x D=64 lanes.
// ---------------------------------------------------------------------------
__device__ __forceinline__ float facet_sum(float v, float* red, int t, int d) {
    __syncthreads();           // protect previous reads of red
    red[t] = v;
    __syncthreads();
    #pragma unroll
    for (int off = 32; off >= 1; off >>= 1) {
        if (d < off) red[t] += red[t + off];
        __syncthreads();
    }
    return red[t & ~63];
}

// HomoAttention, all 8 relations in one launch (grid.y = relation index)
struct HomoJob   { const float* S; const float* Nb; const float* a; float* out; };
struct HomoBatch { HomoJob j[8]; };

__global__ __launch_bounds__(256)
void homo_batch(HomoBatch hb) {
    const HomoJob job = hb.j[blockIdx.y];
    const int b = blockIdx.x, t = threadIdx.x;
    const int d = t & 63;
    __shared__ float red[256];
    const float s = job.S[b * 256 + t];
    const float n = job.Nb[b * 256 + t];
    float sc = facet_sum(s * job.a[d] + n * job.a[64 + d], red, t, d);
    sc = (sc > 0.0f) ? sc : 0.2f * sc;                 // leaky_relu(0.2)
    const float alpha = 1.0f / (1.0f + __expf(-sc));   // sigmoid
    job.out[b * 256 + t] = s + alpha * n;              // safe in-place (out==Nb)
}

// HeteAttention (K=4 channels, NITER=3), user & business sides in one launch
struct HeteJob   { const float* S; const float* Z[4]; float* out; };
struct HeteBatch { HeteJob j[2]; };

__global__ __launch_bounds__(256)
void hete_batch(HeteBatch hb) {
    const HeteJob job = hb.j[blockIdx.y];
    const int b = blockIdx.x, t = threadIdx.x;
    const int d = t & 63;
    __shared__ float red[256];

    const float s = job.S[b * 256 + t];
    float z[4];
    #pragma unroll
    for (int k = 0; k < 4; ++k) z[k] = job.Z[k][b * 256 + t];

    float ss = facet_sum(s * s, red, t, d);
    const float u0 = s / (sqrtf(ss) + 1e-8f);
    #pragma unroll
    for (int k = 0; k < 4; ++k) {
        float zz = facet_sum(z[k] * z[k], red, t, d);
        z[k] = z[k] / (sqrtf(zz) + 1e-8f);
    }

    float u = u0;
    #pragma unroll
    for (int it = 0; it < 3; ++it) {
        float p[4];
        #pragma unroll
        for (int k = 0; k < 4; ++k) p[k] = facet_sum(u * z[k], red, t, d);
        float mx = fmaxf(fmaxf(p[0], p[1]), fmaxf(p[2], p[3]));
        float sum = 0.0f;
        #pragma unroll
        for (int k = 0; k < 4; ++k) { p[k] = __expf(p[k] - mx); sum += p[k]; }
        float nu = u0;
        #pragma unroll
        for (int k = 0; k < 4; ++k) nu += (p[k] / sum) * z[k];
        float ns = facet_sum(nu * nu, red, t, d);
        u = nu / (sqrtf(ns) + 1e-8f);
    }
    job.out[b * 256 + t] = u;
}

// Final: per-facet dot(updated_user, updated_business) -> softmax over F=4
__global__ __launch_bounds__(256)
void ui_softmax(const float* __restrict__ UU, const float* __restrict__ UB,
                float* __restrict__ out) {
    const int b = blockIdx.x, t = threadIdx.x;
    const int f = t >> 6, d = t & 63;
    __shared__ float red[256];
    __shared__ float lf[4];
    float l = facet_sum(UU[b * 256 + t] * UB[b * 256 + t], red, t, d);
    if (d == 0) lf[f] = l;
    __syncthreads();
    if (t < 4) {
        float m = fmaxf(fmaxf(lf[0], lf[1]), fmaxf(lf[2], lf[3]));
        float e0 = __expf(lf[0] - m), e1 = __expf(lf[1] - m);
        float e2 = __expf(lf[2] - m), e3 = __expf(lf[3] - m);
        out[b * 4 + t] = __expf(lf[t] - m) / (e0 + e1 + e2 + e3);
    }
}

// ---------------------------------------------------------------------------
extern "C" void kernel_launch(void* const* d_in, const int* in_sizes, int n_in,
                              void* d_out, int out_size, void* d_ws, size_t ws_size,
                              hipStream_t stream) {
    const float* users       = (const float*)d_in[0];
    const float* businesses  = (const float*)d_in[1];
    const float* un_user     = (const float*)d_in[2];
    const float* un_bus      = (const float*)d_in[3];
    const float* un_city     = (const float*)d_in[4];
    const float* un_cat      = (const float*)d_in[5];
    const float* bn_user     = (const float*)d_in[6];
    const float* bn_bus      = (const float*)d_in[7];
    const float* bn_city     = (const float*)d_in[8];
    const float* bn_cat      = (const float*)d_in[9];
    const float* W_user = (const float*)d_in[10]; const float* b_user = (const float*)d_in[11];
    const float* W_bus  = (const float*)d_in[12]; const float* b_bus  = (const float*)d_in[13];
    const float* W_city = (const float*)d_in[14]; const float* b_city = (const float*)d_in[15];
    const float* W_cat  = (const float*)d_in[16]; const float* b_cat  = (const float*)d_in[17];
    const float* homo_a = (const float*)d_in[18];

    const int B  = out_size / 4;            // 1024
    const int Ku = in_sizes[10] / 256;      // 20000
    const int Kb = in_sizes[12] / 256;      // 20000
    const int Kc = in_sizes[14] / 256;      // 1000
    const int Kg = in_sizes[16] / 256;      // 500
    const int EMB = B * 256;

    float* ws     = (float*)d_ws;
    float* e_user = ws;
    float* e_bus  = ws + (size_t)EMB;
    float* un[4]  = { ws + 2*(size_t)EMB, ws + 3*(size_t)EMB, ws + 4*(size_t)EMB, ws + 5*(size_t)EMB };
    float* bn[4]  = { ws + 6*(size_t)EMB, ws + 7*(size_t)EMB, ws + 8*(size_t)EMB, ws + 9*(size_t)EMB };
    float* uu     = ws + 10*(size_t)EMB;
    float* ub     = ws + 11*(size_t)EMB;

    GemmBatch gb;
    gb.j[0] = { users,      W_user, b_user, e_user, Ku };
    gb.j[1] = { businesses, W_bus,  b_bus,  e_bus,  Kb };
    gb.j[2] = { un_user,    W_user, b_user, un[0],  Ku };
    gb.j[3] = { un_bus,     W_bus,  b_bus,  un[1],  Kb };
    gb.j[4] = { un_city,    W_city, b_city, un[2],  Kc };
    gb.j[5] = { un_cat,     W_cat,  b_cat,  un[3],  Kg };
    gb.j[6] = { bn_user,    W_user, b_user, bn[0],  Ku };
    gb.j[7] = { bn_bus,     W_bus,  b_bus,  bn[1],  Kb };
    gb.j[8] = { bn_city,    W_city, b_city, bn[2],  Kc };
    gb.j[9] = { bn_cat,     W_cat,  b_cat,  bn[3],  Kg };

    embed_gemm<<<dim3(4, B / 64, 10), 128, 0, stream>>>(gb);

    HomoBatch hb;
    for (int i = 0; i < 4; ++i) {
        hb.j[i]     = { e_user, un[i], homo_a + (size_t)i * 128,       un[i] };
        hb.j[4 + i] = { e_bus,  bn[i], homo_a + (size_t)(4 + i) * 128, bn[i] };
    }
    homo_batch<<<dim3(B, 8), 256, 0, stream>>>(hb);

    HeteBatch tb;
    tb.j[0] = { e_user, { un[0], un[1], un[2], un[3] }, uu };
    // faithful to reference: user_embed is the "self" for the business encoder
    tb.j[1] = { e_user, { bn[0], bn[1], bn[2], bn[3] }, ub };
    hete_batch<<<dim3(B, 2), 256, 0, stream>>>(tb);

    ui_softmax<<<B, 256, 0, stream>>>(uu, ub, (float*)d_out);
}
